// Attention_89172110999574
// MI455X (gfx1250) — compile-verified, matrix operand order checked by reading, not run on hardware
//
#include <hip/hip_runtime.h>
#include <math.h>

// ---------- types ----------
typedef unsigned short bfr;                                    // raw bf16 storage
typedef __attribute__((ext_vector_type(16))) __bf16 v16bf;
typedef __attribute__((ext_vector_type(8)))  __bf16 v8bf;
typedef __attribute__((ext_vector_type(8)))  float  v8f;

// ---------- helpers ----------
__device__ __forceinline__ bfr f2br(float f) {
    union { float f; unsigned u; } c; c.f = f;
    unsigned r = c.u + 0x7FFFu + ((c.u >> 16) & 1u);           // round-to-nearest-even
    return (bfr)(r >> 16);
}

// Load a 16-element bf16 WMMA operand for this lane: 8 contiguous + 8 contiguous at +16.
__device__ __forceinline__ v16bf ld_op(const bfr* p) {
    v8bf a0 = *(const v8bf*)(const void*)p;
    v8bf a1 = *(const v8bf*)(const void*)(p + 16);
    v16bf a;
#pragma unroll
    for (int i = 0; i < 8; ++i) { a[i] = a0[i]; a[i + 8] = a1[i]; }
    return a;
}

__device__ __forceinline__ float gelu_exact(float x) {
    return 0.5f * x * (1.0f + erff(x * 0.70710678118654752f));
}

// ---------- constants ----------
#define NPOS   1024          // 32*32 tokens
#define HEADS  8
#define DK     32
#define DV     64
#define DIMIN  256
#define DIMOUT 256
#define INNERV 512           // HEADS*DV
#define QK_SCALE 0.17677669529663687f   // 1/sqrt(32)
#define BN_INV   0.99999500003749969f   // 1/sqrt(1+1e-5)

// ---------- prep: fp32 -> bf16 convert ----------
__global__ void cvt_bf16_kernel(const float* __restrict__ src, bfr* __restrict__ dst, int n) {
    int i = blockIdx.x * blockDim.x + threadIdx.x;
    if (i < n) dst[i] = f2br(src[i]);
}

// dst[n*K + k] = bf16(src[k*N + n])   (transpose to N-major for contiguous-K operand loads)
__global__ void transpose_cvt_kernel(const float* __restrict__ src, bfr* __restrict__ dst,
                                     int K, int N) {
    int i = blockIdx.x * blockDim.x + threadIdx.x;
    if (i >= K * N) return;
    int n = i / K, k = i % K;
    dst[i] = f2br(src[k * N + n]);
}

// ---------- kernel 1: QKV projection GEMM (M=8192, K=256, N=1024), 1 wave = 16x16 tile ----
__global__ __launch_bounds__(128) void qkv_gemm_kernel(
    const bfr* __restrict__ xb,   // (8192, 256) bf16 row-major
    const bfr* __restrict__ WqT,  // (256, 256)  N-major
    const bfr* __restrict__ WkT,  // (256, 256)
    const bfr* __restrict__ WvT,  // (512, 256)
    bfr* __restrict__ Qs,         // (b,h,n,dk) bf16, pre-scaled by 1/sqrt(DK)
    bfr* __restrict__ Kb,         // (b,h,n,dk)
    bfr* __restrict__ Vt)         // (b,h,dv,n)  transposed for attention B loads
{
    const int lane = threadIdx.x & 31;
    const int wv   = threadIdx.x >> 5;
    const int tile = blockIdx.x * 4 + wv;                    // 32768 tiles
    const int tm   = tile >> 6;                              // 0..511 (M tiles)
    const int tn   = tile & 63;                              // 0..63  (N tiles)
    const int hlf  = lane >> 4;
    const int kb   = hlf * 8;
    const int mA   = lane & 15;
    const int nLoc = lane & 15;

    const int cbase = tn * 16;
    const bfr* Bt; int ccol;
    if (cbase < 256)      { Bt = WqT; ccol = cbase;       }
    else if (cbase < 512) { Bt = WkT; ccol = cbase - 256; }
    else                  { Bt = WvT; ccol = cbase - 512; }

    const int rowA = tm * 16 + mA;
    const int nB   = ccol + nLoc;

    v8f acc = {};
#pragma unroll
    for (int k0 = 0; k0 < DIMIN; k0 += 32) {
        v16bf a = ld_op(xb + rowA * DIMIN + k0 + kb);
        v16bf b = ld_op(Bt + nB  * DIMIN + k0 + kb);
        acc = __builtin_amdgcn_wmma_f32_16x16x32_bf16(false, a, false, b,
                                                      (short)0, acc, false, false);
    }

    const int c = cbase + nLoc;                              // global output column 0..1023
#pragma unroll
    for (int r = 0; r < 8; ++r) {
        int m    = tm * 16 + r + hlf * 8;                    // global row 0..8191
        int bidx = m >> 10;
        int npos = m & (NPOS - 1);
        float v  = acc[r];
        if (c < 256) {                                       // Q
            int h = c >> 5, dk = c & 31;
            Qs[(((bidx * HEADS + h) * NPOS + npos) << 5) + dk] = f2br(v * QK_SCALE);
        } else if (c < 512) {                                // K
            int cc = c - 256; int h = cc >> 5, dk = cc & 31;
            Kb[(((bidx * HEADS + h) * NPOS + npos) << 5) + dk] = f2br(v);
        } else {                                             // V (transposed store)
            int cc = c - 512; int h = cc >> 6, dv = cc & 63;
            Vt[(((bidx * HEADS + h) * DV + dv) << 10) + npos] = f2br(v);
        }
    }
}

// ---------- kernel 2: fused flash attention + pos bias + GELU --------------------------
__global__ __launch_bounds__(128) void attn_kernel(
    const bfr*  __restrict__ Qs,
    const bfr*  __restrict__ Kb,
    const bfr*  __restrict__ Vt,
    const float* __restrict__ pos_emb,   // (1024, 8) fp32
    bfr* __restrict__ att)               // (b, n, h*64+dv) bf16 (post-GELU)
{
    __shared__ __attribute__((aligned(16))) bfr ldsP[4][16 * 32];

    const int lane = threadIdx.x & 31;
    const int wv   = threadIdx.x >> 5;
    const int gw   = blockIdx.x * 4 + wv;        // 4096 waves total
    const int qt   = gw & 63;                    // q tile 0..63
    const int bh   = gw >> 6;                    // 0..63
    const int bat  = bh >> 3;
    const int h    = bh & 7;
    const int hlf  = lane >> 4;
    const int kb   = hlf * 8;
    const int mA   = lane & 15;
    const int nLoc = lane & 15;
    const int i0   = qt * 16;

    // Q tile as A operand (held for the whole kernel)
    const v16bf aq = ld_op(Qs + ((bh * NPOS) + i0 + mA) * DK + kb);

    float mrun[8], lrun[8];
    v8f oAcc[4] = { {}, {}, {}, {} };
#pragma unroll
    for (int r = 0; r < 8; ++r) { mrun[r] = -1e30f; lrun[r] = 0.0f; }

    for (int jj = 0; jj < NPOS; jj += 32) {
        // ---- scores: two 16x16 tiles via one WMMA each (K = DK = 32 exactly) ----
        v16bf bk0 = ld_op(Kb + (bh * NPOS + jj +      nLoc) * DK + kb);
        v16bf bk1 = ld_op(Kb + (bh * NPOS + jj + 16 + nLoc) * DK + kb);
        v8f z = {};
        v8f s0 = __builtin_amdgcn_wmma_f32_16x16x32_bf16(false, aq, false, bk0,
                                                         (short)0, z, false, false);
        v8f s1 = __builtin_amdgcn_wmma_f32_16x16x32_bf16(false, aq, false, bk1,
                                                         (short)0, z, false, false);

        float p0[8], p1[8], corr[8];
#pragma unroll
        for (int r = 0; r < 8; ++r) {
            const int i  = i0 + r + hlf * 8;
            const int j0 = jj + nLoc, j1 = j0 + 16;
            const int ri = i >> 5,  ci = i & 31;
            int d00 = ri - (j0 >> 5); d00 = d00 < 0 ? -d00 : d00;
            int d01 = ci - (j0 & 31); d01 = d01 < 0 ? -d01 : d01;
            int d10 = ri - (j1 >> 5); d10 = d10 < 0 ? -d10 : d10;
            int d11 = ci - (j1 & 31); d11 = d11 < 0 ? -d11 : d11;
            float v0 = s0[r] + pos_emb[((d00 << 5) + d01) * HEADS + h];
            float v1 = s1[r] + pos_emb[((d10 << 5) + d11) * HEADS + h];

            float mx = fmaxf(v0, v1);
            mx = fmaxf(mx, __shfl_xor(mx, 1, 16));
            mx = fmaxf(mx, __shfl_xor(mx, 2, 16));
            mx = fmaxf(mx, __shfl_xor(mx, 4, 16));
            mx = fmaxf(mx, __shfl_xor(mx, 8, 16));

            float mnew = fmaxf(mrun[r], mx);
            float cfac = __expf(mrun[r] - mnew);
            mrun[r] = mnew;
            v0 = __expf(v0 - mnew);
            v1 = __expf(v1 - mnew);
            float rs = v0 + v1;
            rs += __shfl_xor(rs, 1, 16);
            rs += __shfl_xor(rs, 2, 16);
            rs += __shfl_xor(rs, 4, 16);
            rs += __shfl_xor(rs, 8, 16);
            lrun[r] = lrun[r] * cfac + rs;
            corr[r] = cfac;
            p0[r] = v0; p1[r] = v1;
        }
#pragma unroll
        for (int r = 0; r < 8; ++r) {
            oAcc[0][r] *= corr[r]; oAcc[1][r] *= corr[r];
            oAcc[2][r] *= corr[r]; oAcc[3][r] *= corr[r];
        }

        // ---- re-layout P (C/D layout) -> A-operand layout through LDS ----
        __syncthreads();
#pragma unroll
        for (int r = 0; r < 8; ++r) {
            int m = r + hlf * 8;
            ldsP[wv][m * 32 + nLoc]      = f2br(p0[r]);
            ldsP[wv][m * 32 + nLoc + 16] = f2br(p1[r]);
        }
        __syncthreads();
        v16bf ap = ld_op(&ldsP[wv][mA * 32 + kb]);

        // ---- P (16x32) x V (32x64): 4 WMMAs (V pre-transposed -> contiguous loads) ----
#pragma unroll
        for (int t = 0; t < 4; ++t) {
            v16bf bv = ld_op(Vt + ((bh * DV) + t * 16 + nLoc) * NPOS + jj + kb);
            oAcc[t] = __builtin_amdgcn_wmma_f32_16x16x32_bf16(false, ap, false, bv,
                                                              (short)0, oAcc[t], false, false);
        }
    }

    // ---- normalize, GELU, store bf16 ----
#pragma unroll
    for (int r = 0; r < 8; ++r) {
        float inv = 1.0f / lrun[r];
        int i = i0 + r + hlf * 8;
#pragma unroll
        for (int t = 0; t < 4; ++t) {
            float g = gelu_exact(oAcc[t][r] * inv);
            att[(bat * NPOS + i) * INNERV + h * DV + t * 16 + nLoc] = f2br(g);
        }
    }
}

// ---------- kernel 3: output GEMM (M=8192, K=512, N=256) + bias + BN affine ------------
__global__ __launch_bounds__(128) void out_gemm_kernel(
    const bfr*  __restrict__ att,   // (8192, 512) bf16
    const bfr*  __restrict__ WoT,   // (256, 512)  N-major bf16
    const float* __restrict__ bo,
    const float* __restrict__ gamma,
    const float* __restrict__ beta,
    float* __restrict__ out)        // (8192, 256) fp32
{
    const int lane = threadIdx.x & 31;
    const int wv   = threadIdx.x >> 5;
    const int tile = blockIdx.x * 4 + wv;        // 8192 tiles
    const int tm   = tile >> 4;                  // 0..511
    const int tn   = tile & 15;                  // 0..15
    const int hlf  = lane >> 4;
    const int kb   = hlf * 8;
    const int mA   = lane & 15;
    const int nLoc = lane & 15;

    const int rowA = tm * 16 + mA;
    const int nB   = tn * 16 + nLoc;

    v8f acc = {};
#pragma unroll
    for (int k0 = 0; k0 < INNERV; k0 += 32) {
        v16bf a = ld_op(att + rowA * INNERV + k0 + kb);
        v16bf b = ld_op(WoT + nB  * INNERV + k0 + kb);
        acc = __builtin_amdgcn_wmma_f32_16x16x32_bf16(false, a, false, b,
                                                      (short)0, acc, false, false);
    }

    const int n = tn * 16 + nLoc;
    const float gm = gamma[n] * BN_INV;
    const float bt = beta[n];
    const float bb = bo[n];
#pragma unroll
    for (int r = 0; r < 8; ++r) {
        int m = tm * 16 + r + hlf * 8;
        out[m * DIMOUT + n] = gm * (acc[r] + bb) + bt;
    }
}

// ---------- host launcher ----------
extern "C" void kernel_launch(void* const* d_in, const int* in_sizes, int n_in,
                              void* d_out, int out_size, void* d_ws, size_t ws_size,
                              hipStream_t stream) {
    const float* x       = (const float*)d_in[0];   // (8,32,32,256)
    const float* Wq      = (const float*)d_in[1];   // (256,256)
    const float* Wk      = (const float*)d_in[2];   // (256,256)
    const float* Wv      = (const float*)d_in[3];   // (256,512)
    const float* pos_emb = (const float*)d_in[4];   // (1024,8)
    const float* Wo      = (const float*)d_in[5];   // (512,256)
    const float* bo      = (const float*)d_in[6];
    const float* gamma   = (const float*)d_in[7];
    const float* beta    = (const float*)d_in[8];
    // d_in[9] = pos_indices (recomputed analytically on device, unused)

    // workspace carve-out (all offsets 256B aligned -> 16B operand alignment holds)
    size_t off = 0;
    auto carve = [&](size_t bytes) {
        void* p = (char*)d_ws + off;
        off += (bytes + 255) & ~(size_t)255;
        return p;
    };
    bfr* xb  = (bfr*)carve((size_t)8192 * 256 * 2);
    bfr* WqT = (bfr*)carve((size_t)256 * 256 * 2);
    bfr* WkT = (bfr*)carve((size_t)256 * 256 * 2);
    bfr* WvT = (bfr*)carve((size_t)512 * 256 * 2);
    bfr* WoT = (bfr*)carve((size_t)256 * 512 * 2);
    bfr* Qs  = (bfr*)carve((size_t)64 * 1024 * 32 * 2);
    bfr* Kb  = (bfr*)carve((size_t)64 * 1024 * 32 * 2);
    bfr* Vt  = (bfr*)carve((size_t)64 * 64 * 1024 * 2);
    bfr* att = (bfr*)carve((size_t)8192 * 512 * 2);

    // prep: bf16 conversion + weight transposes
    cvt_bf16_kernel<<<(2097152 + 255) / 256, 256, 0, stream>>>(x, xb, 2097152);
    transpose_cvt_kernel<<<(65536  + 255) / 256, 256, 0, stream>>>(Wq, WqT, 256, 256);
    transpose_cvt_kernel<<<(65536  + 255) / 256, 256, 0, stream>>>(Wk, WkT, 256, 256);
    transpose_cvt_kernel<<<(131072 + 255) / 256, 256, 0, stream>>>(Wv, WvT, 256, 512);
    transpose_cvt_kernel<<<(131072 + 255) / 256, 256, 0, stream>>>(Wo, WoT, 512, 256);

    // QKV projection: 512 x 64 tiles, 4 waves/block
    qkv_gemm_kernel<<<8192, 128, 0, stream>>>(xb, WqT, WkT, WvT, Qs, Kb, Vt);

    // fused attention: 64 (b,h) x 64 q-tiles = 4096 waves
    attn_kernel<<<1024, 128, 0, stream>>>(Qs, Kb, Vt, pos_emb, att);

    // output projection + bias + BN affine
    out_gemm_kernel<<<2048, 128, 0, stream>>>(att, WoT, bo, gamma, beta, (float*)d_out);
}
